// MatryoshkaTranscoder_21303037788826
// MI455X (gfx1250) — compile-verified, matrix operand order checked by reading, not run on hardware
//
#include <hip/hip_runtime.h>
#include <hip/hip_bf16.h>

typedef __attribute__((ext_vector_type(16))) __bf16 v16bf;
typedef __attribute__((ext_vector_type(8)))  float  v8f;

#define TM 128
#define TN 128
#define TK 32
#define ROWF 36            // 32 f32 + 4 pad floats -> 144B row stride (16B aligned,
                           // 36*4/4 = 36 banks/row, gcd(36,64)=4 -> 16 distinct banks)
#define TILE_FLOATS (128 * ROWF)       // 4608 floats = 18432 B per tile buffer

// LDS byte offset of a generic pointer into __shared__ (addrspacecast -> ptrtoint)
__device__ __forceinline__ unsigned int lds_off_of(const void* p) {
  return (unsigned int)(unsigned long long)(__attribute__((address_space(3))) const void*)p;
}

// Load one 16x32-bf16 fragment row-slice from an f32 LDS row, converting to bf16.
// Lane layout per ISA 16-bit A/B fragments: half==0 -> K {0..7,16..23},
// half==1 -> K {8..15,24..31}.
__device__ __forceinline__ v16bf load_frag_cvt(const float* rowp, int half) {
  const float4* p0 = (const float4*)(rowp + half * 8);
  const float4* p1 = (const float4*)(rowp + 16 + half * 8);
  float4 a = p0[0], b = p0[1], c = p1[0], d = p1[1];
  v16bf r;
  r[0] = (__bf16)a.x;  r[1] = (__bf16)a.y;  r[2] = (__bf16)a.z;  r[3] = (__bf16)a.w;
  r[4] = (__bf16)b.x;  r[5] = (__bf16)b.y;  r[6] = (__bf16)b.z;  r[7] = (__bf16)b.w;
  r[8] = (__bf16)c.x;  r[9] = (__bf16)c.y;  r[10] = (__bf16)c.z; r[11] = (__bf16)c.w;
  r[12] = (__bf16)d.x; r[13] = (__bf16)d.y; r[14] = (__bf16)d.z; r[15] = (__bf16)d.w;
  return r;
}

// ---------------------------------------------------------------------------
// Phase 1: pre = h2 @ W^T + b ; z = relu(pre) + (pre > 1) ; z -> Z (f32, NT)
// Block = 256 threads = 8 wave32 arranged 4(M) x 2(N); wave tile 32x64 =
// 2x4 v_wmma_f32_16x16x32_bf16. K-tiles (f32) stream into double-buffered LDS
// via global_load_async_to_lds_b128 (ASYNCcnt), f32->bf16 cvt at fragment load.
// Dynamic LDS layout (floats): [A0 | A1 | B0 | B1], each TILE_FLOATS.
// ---------------------------------------------------------------------------
__global__ __launch_bounds__(256) void encoder_gemm_jumprelu(
    const float* __restrict__ A,     // [M,K] h_2
    const float* __restrict__ W,     // [N,K] W_enc
    const float* __restrict__ bias,  // [N]
    float* __restrict__ Z,           // [M,N]
    int M, int N, int K)
{
  extern __shared__ __attribute__((aligned(16))) float smemf[];

  const int tid  = threadIdx.x;
  const int lane = tid & 31;
  const int wave = tid >> 5;     // 0..7
  const int wm   = wave >> 1;    // 0..3  (M direction, 32 rows each)
  const int wn   = wave & 1;     // 0..1  (N direction, 64 cols each)
  const int m0   = blockIdx.y * TM;
  const int n0   = blockIdx.x * TN;
  const int mrow = lane & 15;
  const int half = lane >> 4;

  v8f acc[2][4] = {};            // [mi][ni] 16x16 f32 accumulators

  // LDS byte bases of the four tile buffers
  const unsigned int ldsBase = lds_off_of(smemf);
  const unsigned int aBase[2] = { ldsBase,
                                  ldsBase + TILE_FLOATS * 4u };
  const unsigned int bBase[2] = { ldsBase + 2u * TILE_FLOATS * 4u,
                                  ldsBase + 3u * TILE_FLOATS * 4u };

  // Per-thread async-copy chunk geometry: both tiles are 128 rows x 8 x b128
  // chunks = 1024 chunks; each thread moves 4 chunks per tile per K-step.
  unsigned int ldsChunk[4], aByte[4], bByte[4];
#pragma unroll
  for (int i = 0; i < 4; ++i) {
    const int c = tid + i * 256;                  // chunk id 0..1023
    const int r = c >> 3;                         // tile row 0..127
    const unsigned int colb = (unsigned int)(c & 7) * 16u;  // byte col in row
    ldsChunk[i] = (unsigned int)r * (ROWF * 4u) + colb;
    aByte[i] = ((unsigned int)(m0 + r) * (unsigned int)K) * 4u + colb;
    bByte[i] = ((unsigned int)(n0 + r) * (unsigned int)K) * 4u + colb;
  }

  // async copy of one K-step's A+B tiles into buffer b (8 x b128 per thread)
  auto issue_group = [&](int k0, int b) {
    const unsigned int kb = (unsigned int)k0 * 4u;
#pragma unroll
    for (int i = 0; i < 4; ++i)
      asm volatile("global_load_async_to_lds_b128 %0, %1, %2"
                   :: "v"(aBase[b] + ldsChunk[i]), "v"(aByte[i] + kb), "s"(A)
                   : "memory");
#pragma unroll
    for (int i = 0; i < 4; ++i)
      asm volatile("global_load_async_to_lds_b128 %0, %1, %2"
                   :: "v"(bBase[b] + ldsChunk[i]), "v"(bByte[i] + kb), "s"(W)
                   : "memory");
  };

  issue_group(0, 0);                              // prime buffer 0

  const int steps = K / TK;
  for (int s = 0; s < steps; ++s) {
    const int b = s & 1;

    // my async writes for buffer b are done; barrier publishes all waves'
    asm volatile("s_wait_asynccnt 0x0" ::: "memory");
    __syncthreads();

    // overlap: stream next K-tile into the other buffer while we compute.
    // Safe: buffer b^1 was last read in iteration s-1, and every wave
    // consumed (dscnt-waited) those reads before reaching the barrier above.
    if (s + 1 < steps) issue_group((s + 1) * TK, b ^ 1);

    const float* Ab = smemf + b * TILE_FLOATS;
    const float* Bb = smemf + 2 * TILE_FLOATS + b * TILE_FLOATS;

    v16bf afrag[2], bfrag[4];
#pragma unroll
    for (int mi = 0; mi < 2; ++mi)
      afrag[mi] = load_frag_cvt(Ab + (wm * 32 + mi * 16 + mrow) * ROWF, half);
#pragma unroll
    for (int ni = 0; ni < 4; ++ni)
      bfrag[ni] = load_frag_cvt(Bb + (wn * 64 + ni * 16 + mrow) * ROWF, half);

#pragma unroll
    for (int mi = 0; mi < 2; ++mi)
#pragma unroll
      for (int ni = 0; ni < 4; ++ni)
        acc[mi][ni] = __builtin_amdgcn_wmma_f32_16x16x32_bf16(
            false, afrag[mi], false, bfrag[ni],
            (short)0, acc[mi][ni], false, false);
  }

  // epilogue: bias + JumpReLU (gamma=1, beta=1). C/D layout: lane -> N =
  // lane&15, VGPR r -> M = r + 8*(lane>=16). Non-temporal stores: z (512 MB)
  // can never be L2-resident; keep W_enc (128 MB) hot in the 192 MB L2.
#pragma unroll
  for (int ni = 0; ni < 4; ++ni) {
    const int gcol = n0 + wn * 64 + ni * 16 + mrow;
    const float bv = bias[gcol];
#pragma unroll
    for (int mi = 0; mi < 2; ++mi) {
#pragma unroll
      for (int r = 0; r < 8; ++r) {
        const int grow = m0 + wm * 32 + mi * 16 + r + half * 8;
        const float pre = acc[mi][ni][r] + bv;
        const float z = fmaxf(pre, 0.0f) + ((pre > 1.0f) ? 1.0f : 0.0f);
        __builtin_nontemporal_store(z, &Z[(size_t)grow * N + gcol]);
      }
    }
  }
}

// ---------------------------------------------------------------------------
// Phase 2: exact top-k mask per (row, slice). z >= 0 so f32 bits are
// order-monotone; 4-round 8-bit radix select over the LDS-cached slice
// recovers the exact k-th largest value; entries strictly below are zeroed.
// ---------------------------------------------------------------------------
__global__ __launch_bounds__(256) void topk_mask_radix(
    float* __restrict__ Z, int rowLen, int sliceOff, int sliceLen, int k)
{
  extern __shared__ unsigned int vals[];          // [sliceLen] z bit patterns
  __shared__ unsigned int hist[256];
  __shared__ unsigned int s_prefix;
  __shared__ unsigned int s_need;

  const int tid = threadIdx.x;
  float* zrow = Z + (size_t)blockIdx.x * rowLen + sliceOff;

  for (int i = tid; i < sliceLen; i += blockDim.x)
    vals[i] = __float_as_uint(__builtin_nontemporal_load(&zrow[i]));
  if (tid == 0) { s_prefix = 0u; s_need = (unsigned)k; }
  __syncthreads();

  for (int round = 0; round < 4; ++round) {
    const int shift = 24 - round * 8;
    for (int i = tid; i < 256; i += blockDim.x) hist[i] = 0u;
    __syncthreads();

    const unsigned int pmask = (round == 0) ? 0u : (0xFFFFFFFFu << (shift + 8));
    const unsigned int pref  = s_prefix;
    for (int i = tid; i < sliceLen; i += blockDim.x) {
      const unsigned int v = vals[i];
      if ((v & pmask) == pref)
        atomicAdd(&hist[(v >> shift) & 0xFFu], 1u);
    }
    __syncthreads();

    if (tid == 0) {
      unsigned int need = s_need, cum = 0u;
      int b = 255;
      for (; b > 0; --b) {
        if (cum + hist[b] >= need) break;
        cum += hist[b];
      }
      s_need   = need - cum;
      s_prefix = pref | ((unsigned int)b << shift);
    }
    __syncthreads();
  }

  const unsigned int thr = s_prefix;   // exact bits of k-th largest value
  for (int i = tid; i < sliceLen; i += blockDim.x)
    if (vals[i] < thr) zrow[i] = 0.0f;
}

// ---------------------------------------------------------------------------
extern "C" void kernel_launch(void* const* d_in, const int* in_sizes, int n_in,
                              void* d_out, int out_size, void* d_ws, size_t ws_size,
                              hipStream_t stream) {
  (void)in_sizes; (void)n_in; (void)out_size; (void)d_ws; (void)ws_size;

  const float* h2 = (const float*)d_in[0];   // [4096,1024]
  const float* W  = (const float*)d_in[1];   // [32768,1024]
  const float* b  = (const float*)d_in[2];   // [32768]
  float* out      = (float*)d_out;           // [4096,32768]

  const int M = 4096, N = 32768, K = 1024;

  dim3 grid(N / TN, M / TM);                 // (256, 32)
  const size_t gemmLds = 4u * TILE_FLOATS * 4u;   // 72 KB, double-buffered A/B
  encoder_gemm_jumprelu<<<grid, dim3(256), gemmLds, stream>>>(h2, W, b, out, M, N, K);

  // nested levels: [0,2048) k=32 ; [2048,8192) k=64 ; [8192,32768) k=128
  topk_mask_radix<<<M, 256, (size_t)2048  * 4, stream>>>(out, N, 0,    2048,  32);
  topk_mask_radix<<<M, 256, (size_t)6144  * 4, stream>>>(out, N, 2048, 6144,  64);
  topk_mask_radix<<<M, 256, (size_t)24576 * 4, stream>>>(out, N, 8192, 24576, 128);
}
